// MoEBlock_51032801411144
// MI455X (gfx1250) — compile-verified, matrix operand order checked by reading, not run on hardware
//
#include <hip/hip_runtime.h>
#include <math.h>

// Problem constants (reference setup): B=4,S=2048 -> T=8192, D=1024, H=4096, E=8, top_k=2
#define T_TOK 8192
#define DDIM  1024
#define HDIM  4096
#define NEXP  8
#define TOPK  2

// GEMM tiles: 256 threads = 8 wave32, 128x128 block tile, K-step 32 (bf16 WMMA)
#define BM 128
#define BN 128
#define BK 32

typedef __attribute__((ext_vector_type(16))) __bf16 v16bf;
typedef __attribute__((ext_vector_type(8)))  float  v8f;
typedef __attribute__((ext_vector_type(4)))  __bf16 bf16x4;

union FragBF { v16bf v; __bf16 e[16]; };
union Acc8   { v8f   v; float  f[8];  };

// ---- CDNA5 async global->LDS path (guarded; fallback = VGPR-mediated copy) ----
#if defined(__gfx1250__)
# if __has_builtin(__builtin_amdgcn_global_load_async_to_lds_b128) && \
     __has_builtin(__builtin_amdgcn_s_wait_asynccnt)
#  define USE_ASYNC 1
# endif
#endif
#ifndef USE_ASYNC
# define USE_ASYNC 0
#endif

// builtin's parameter types (from hipcc diagnostic):
//   param0: int __vector_size__(16) __device__(AS1) *   (global src)
//   param1: LDS dst -> analogous AS3 typed pointer
typedef int v4i __attribute__((vector_size(16)));
typedef __attribute__((address_space(1))) v4i* gv4i_p;
typedef __attribute__((address_space(3))) v4i* lv4i_p;

// copy 32 contiguous bytes global -> LDS (both 16B aligned)
__device__ __forceinline__ void cp32B(void* lds_dst, const void* gsrc) {
#if USE_ASYNC
    __builtin_amdgcn_global_load_async_to_lds_b128(
        (gv4i_p)(void*)gsrc, (lv4i_p)lds_dst, 0, 0);
    __builtin_amdgcn_global_load_async_to_lds_b128(
        (gv4i_p)(void*)((const char*)gsrc + 16), (lv4i_p)((char*)lds_dst + 16), 0, 0);
#else
    const uint4* s = (const uint4*)gsrc;
    uint4* d = (uint4*)lds_dst;
    d[0] = s[0];
    d[1] = s[1];
#endif
}

__device__ __forceinline__ void stage_wait() {
#if USE_ASYNC
    __builtin_amdgcn_s_wait_asynccnt(0);
#endif
    __syncthreads();
}

// ---------------------------------------------------------------------------
// Kernel 0: zero output accumulator + router counters
// ---------------------------------------------------------------------------
__global__ void moe_zero(float* __restrict__ out, long n, int* cnt, float* psum) {
    long i = (long)blockIdx.x * blockDim.x + threadIdx.x;
    long stride = (long)gridDim.x * blockDim.x;
    for (; i < n; i += stride) out[i] = 0.0f;
    if (blockIdx.x == 0 && threadIdx.x < NEXP) {
        cnt[threadIdx.x] = 0;
        psum[threadIdx.x] = 0.0f;
    }
}

// ---------------------------------------------------------------------------
// Kernel 1: fp32 -> bf16 conversion (x, w1, w2), vectorized 4-wide
// ---------------------------------------------------------------------------
__global__ void moe_cvt(const float* __restrict__ src, __bf16* __restrict__ dst, long n4) {
    long i = (long)blockIdx.x * blockDim.x + threadIdx.x;
    long stride = (long)gridDim.x * blockDim.x;
    for (; i < n4; i += stride) {
        float4 v = ((const float4*)src)[i];
        bf16x4 o = { (__bf16)v.x, (__bf16)v.y, (__bf16)v.z, (__bf16)v.w };
        ((bf16x4*)dst)[i] = o;
    }
}

// ---------------------------------------------------------------------------
// Kernel 2: router. One wave32 per token.
// ---------------------------------------------------------------------------
__global__ __launch_bounds__(256)
void moe_router(const float* __restrict__ x, const float* __restrict__ rw,
                int* cnt, float* psum, int* list, float* glist) {
    int gid  = blockIdx.x * blockDim.x + threadIdx.x;
    int tok  = gid >> 5;
    int lane = threadIdx.x & 31;
    if (tok >= T_TOK) return;
    const float* xr = x + (size_t)tok * DDIM;

    float acc[NEXP];
#pragma unroll
    for (int e = 0; e < NEXP; ++e) acc[e] = 0.0f;
    for (int i = 0; i < DDIM / 32; ++i) {
        float xv = xr[i * 32 + lane];
#pragma unroll
        for (int e = 0; e < NEXP; ++e)
            acc[e] += xv * rw[e * DDIM + i * 32 + lane];
    }
#pragma unroll
    for (int off = 16; off > 0; off >>= 1) {
#pragma unroll
        for (int e = 0; e < NEXP; ++e)
            acc[e] += __shfl_xor(acc[e], off, 32);
    }
    if (lane == 0) {
        float mx = acc[0];
#pragma unroll
        for (int e = 1; e < NEXP; ++e) mx = fmaxf(mx, acc[e]);
        float p[NEXP], s = 0.0f;
#pragma unroll
        for (int e = 0; e < NEXP; ++e) { p[e] = __expf(acc[e] - mx); s += p[e]; }
        float inv = 1.0f / s;
#pragma unroll
        for (int e = 0; e < NEXP; ++e) {
            p[e] *= inv;
            atomicAdd(psum + e, p[e]);
        }
        int e0 = 0;
#pragma unroll
        for (int e = 1; e < NEXP; ++e) if (p[e] > p[e0]) e0 = e;
        int e1 = (e0 == 0) ? 1 : 0;
#pragma unroll
        for (int e = 0; e < NEXP; ++e) if (e != e0 && p[e] > p[e1]) e1 = e;
        float g = 1.0f / (p[e0] + p[e1]);
        int pos0 = atomicAdd(cnt + e0, 1);
        list[e0 * T_TOK + pos0]  = tok;
        glist[e0 * T_TOK + pos0] = p[e0] * g;
        int pos1 = atomicAdd(cnt + e1, 1);
        list[e1 * T_TOK + pos1]  = tok;
        glist[e1 * T_TOK + pos1] = p[e1] * g;
    }
}

// ---------------------------------------------------------------------------
// Kernel 3: prefix offsets + aux loss scalar
// ---------------------------------------------------------------------------
__global__ void moe_aux(const int* cnt, const float* psum, int* offs, float* aux_out) {
    if (blockIdx.x == 0 && threadIdx.x == 0) {
        int run = 0; float aux = 0.0f;
        for (int e = 0; e < NEXP; ++e) {
            offs[e] = run; run += cnt[e];
            float fi = (float)cnt[e] / (float)(T_TOK * TOPK);
            float Pi = psum[e] / (float)T_TOK;
            aux += fi * Pi;
        }
        *aux_out = (float)NEXP * aux;
    }
}

// ---------------------------------------------------------------------------
// Kernel 4: layer 1 — h = gelu(xb[list] @ w1b[e] + b1[e])
// Double-buffered LDS, async global->LDS staging, bf16 WMMA.
// Grid: (H/BN, T/BM, E)
// ---------------------------------------------------------------------------
__global__ __launch_bounds__(256)
void moe_ffn1(const __bf16* __restrict__ xb,
              const __bf16* __restrict__ w1b, const float* __restrict__ b1,
              const int* __restrict__ cnt, const int* __restrict__ offs,
              const int* __restrict__ list, __bf16* __restrict__ hbuf) {
    int e  = blockIdx.z;
    int m0 = blockIdx.y * BM;
    int n0 = blockIdx.x * BN;
    int rows = cnt[e];
    if (m0 >= rows) return;
    int rowOff = offs[e];
    const __bf16* w1e = w1b + (size_t)e * DDIM * HDIM;

    __shared__ __bf16 sA[2][BM][BK];   // row-major, 64B rows
    __shared__ __bf16 sB[2][BK][BN];   // K-major (memory order), 256B rows
    __shared__ int    sTok[BM];

    int tid = threadIdx.x;
    if (tid < BM) {
        int r = m0 + tid;
        int rc = (r < rows) ? r : (rows - 1);   // clamp: garbage rows masked at epilogue
        sTok[tid] = list[e * T_TOK + rc];
    }
    __syncthreads();

    int lane = tid & 31;
    int wid  = tid >> 5;
    int wm = wid >> 2;             // 0..1 -> M offset wm*64
    int wn = wid & 3;              // 0..3 -> N offset wn*32
    int lr = lane & 15;
    int kh = (lane >> 4) * 8;      // K-half select (16-bit WMMA operand layout)

    // staging coords
    int arow = tid >> 1, apart = tid & 1;          // A: 2 thr/row, 32B each
    int brow = tid >> 3, bseg  = tid & 7;          // B: 8 thr/row, 32B each
    int atok = sTok[arow];

    v8f zero = {0.f,0.f,0.f,0.f,0.f,0.f,0.f,0.f};
    v8f accv[4][2];
#pragma unroll
    for (int m = 0; m < 4; ++m)
#pragma unroll
        for (int n = 0; n < 2; ++n) accv[m][n] = zero;

    // prologue: fill buffer 0 for k0 = 0
    cp32B(&sA[0][arow][apart * 16], xb + (size_t)atok * DDIM + apart * 16);
    cp32B(&sB[0][brow][bseg * 16],  w1e + (size_t)brow * HDIM + n0 + bseg * 16);
    stage_wait();

    int buf = 0;
    for (int k0 = 0; k0 < DDIM; k0 += BK) {
        int nb = buf ^ 1;
        if (k0 + BK < DDIM) {   // prefetch next K-slab while computing this one
            cp32B(&sA[nb][arow][apart * 16],
                  xb + (size_t)atok * DDIM + (k0 + BK) + apart * 16);
            cp32B(&sB[nb][brow][bseg * 16],
                  w1e + (size_t)(k0 + BK + brow) * HDIM + n0 + bseg * 16);
        }

        FragBF a[4], b[2];
#pragma unroll
        for (int m = 0; m < 4; ++m) {
            int r = wm * 64 + m * 16 + lr;
#pragma unroll
            for (int i = 0; i < 8; ++i) {
                a[m].e[i]     = sA[buf][r][kh + i];
                a[m].e[i + 8] = sA[buf][r][kh + 16 + i];
            }
        }
#pragma unroll
        for (int n = 0; n < 2; ++n) {
            int c = wn * 32 + n * 16 + lr;
#pragma unroll
            for (int i = 0; i < 8; ++i) {
                b[n].e[i]     = sB[buf][kh + i][c];        // transpose at read
                b[n].e[i + 8] = sB[buf][kh + 16 + i][c];
            }
        }
#pragma unroll
        for (int m = 0; m < 4; ++m)
#pragma unroll
            for (int n = 0; n < 2; ++n)
                accv[m][n] = __builtin_amdgcn_wmma_f32_16x16x32_bf16(
                    false, a[m].v, false, b[n].v, (short)0, accv[m][n], false, false);

        stage_wait();           // async done + all waves finished reading `buf`
        buf = nb;
    }

    // epilogue: exact GELU(acc + b1) -> bf16 h
#pragma unroll
    for (int m = 0; m < 4; ++m) {
#pragma unroll
        for (int n = 0; n < 2; ++n) {
            Acc8 u; u.v = accv[m][n];
            int col = n0 + wn * 32 + n * 16 + lr;
            float bias = b1[e * HDIM + col];
#pragma unroll
            for (int i = 0; i < 8; ++i) {
                int mloc = wm * 64 + m * 16 + ((lane >> 4) * 8) + i;
                int mg = m0 + mloc;
                if (mg < rows) {
                    float v = u.f[i] + bias;
                    float g = 0.5f * v * (1.0f + erff(v * 0.70710678118654752f));
                    hbuf[(size_t)(rowOff + mg) * HDIM + col] = (__bf16)g;
                }
            }
        }
    }
}

// ---------------------------------------------------------------------------
// Kernel 5: layer 2 — out[tok] += gate * (h @ w2b[e] + b2[e]) (atomic scatter)
// Grid: (D/BN, T/BM, E)
// ---------------------------------------------------------------------------
__global__ __launch_bounds__(256)
void moe_ffn2(const __bf16* __restrict__ hbuf,
              const __bf16* __restrict__ w2b, const float* __restrict__ b2,
              const int* __restrict__ cnt, const int* __restrict__ offs,
              const int* __restrict__ list, const float* __restrict__ glist,
              float* __restrict__ out) {
    int e  = blockIdx.z;
    int m0 = blockIdx.y * BM;
    int n0 = blockIdx.x * BN;
    int rows = cnt[e];
    if (m0 >= rows) return;
    int rowOff = offs[e];
    const __bf16* w2e = w2b + (size_t)e * HDIM * DDIM;

    __shared__ __bf16 sA[2][BM][BK];
    __shared__ __bf16 sB[2][BK][BN];
    __shared__ int    sTok[BM];
    __shared__ float  sGate[BM];

    int tid = threadIdx.x;
    if (tid < BM) {
        int r = m0 + tid;
        int rc = (r < rows) ? r : (rows - 1);
        sTok[tid]  = list[e * T_TOK + rc];
        sGate[tid] = (r < rows) ? glist[e * T_TOK + rc] : 0.0f;
    }
    __syncthreads();

    int lane = tid & 31;
    int wid  = tid >> 5;
    int wm = wid >> 2;
    int wn = wid & 3;
    int lr = lane & 15;
    int kh = (lane >> 4) * 8;

    int arow = tid >> 1, apart = tid & 1;
    int brow = tid >> 3, bseg  = tid & 7;
    int amg = m0 + arow;
    size_t ahrow = (size_t)(rowOff + ((amg < rows) ? amg : (rows - 1)));

    v8f zero = {0.f,0.f,0.f,0.f,0.f,0.f,0.f,0.f};
    v8f accv[4][2];
#pragma unroll
    for (int m = 0; m < 4; ++m)
#pragma unroll
        for (int n = 0; n < 2; ++n) accv[m][n] = zero;

    cp32B(&sA[0][arow][apart * 16], hbuf + ahrow * HDIM + apart * 16);
    cp32B(&sB[0][brow][bseg * 16],  w2e + (size_t)brow * DDIM + n0 + bseg * 16);
    stage_wait();

    int buf = 0;
    for (int k0 = 0; k0 < HDIM; k0 += BK) {
        int nb = buf ^ 1;
        if (k0 + BK < HDIM) {
            cp32B(&sA[nb][arow][apart * 16],
                  hbuf + ahrow * HDIM + (k0 + BK) + apart * 16);
            cp32B(&sB[nb][brow][bseg * 16],
                  w2e + (size_t)(k0 + BK + brow) * DDIM + n0 + bseg * 16);
        }

        FragBF a[4], b[2];
#pragma unroll
        for (int m = 0; m < 4; ++m) {
            int r = wm * 64 + m * 16 + lr;
#pragma unroll
            for (int i = 0; i < 8; ++i) {
                a[m].e[i]     = sA[buf][r][kh + i];
                a[m].e[i + 8] = sA[buf][r][kh + 16 + i];
            }
        }
#pragma unroll
        for (int n = 0; n < 2; ++n) {
            int c = wn * 32 + n * 16 + lr;
#pragma unroll
            for (int i = 0; i < 8; ++i) {
                b[n].e[i]     = sB[buf][kh + i][c];
                b[n].e[i + 8] = sB[buf][kh + 16 + i][c];
            }
        }
#pragma unroll
        for (int m = 0; m < 4; ++m)
#pragma unroll
            for (int n = 0; n < 2; ++n)
                accv[m][n] = __builtin_amdgcn_wmma_f32_16x16x32_bf16(
                    false, a[m].v, false, b[n].v, (short)0, accv[m][n], false, false);

        stage_wait();
        buf = nb;
    }

#pragma unroll
    for (int m = 0; m < 4; ++m) {
#pragma unroll
        for (int n = 0; n < 2; ++n) {
            Acc8 u; u.v = accv[m][n];
            int col = n0 + wn * 32 + n * 16 + lr;
            float bias = b2[e * DDIM + col];
#pragma unroll
            for (int i = 0; i < 8; ++i) {
                int mloc = wm * 64 + m * 16 + ((lane >> 4) * 8) + i;
                int mg = m0 + mloc;
                if (mg < rows) {
                    float v = (u.f[i] + bias) * sGate[mloc];
                    atomicAdd(out + (size_t)sTok[mloc] * DDIM + col, v);
                }
            }
        }
    }
}

// ---------------------------------------------------------------------------
// Workspace layout (~273 MB):
//   [0..32)   cnt[8]   [32..64) offs[8]   [64..96) psum[8]
//   [256..)          list  (E*T int,  256 KiB)
//   [+256KiB..)      glist (E*T f32,  256 KiB)
//   [1MiB..)         xb    (T*D bf16,   16 MiB)
//   [17MiB..)        w1b   (E*D*H bf16, 64 MiB)
//   [81MiB..)        w2b   (E*H*D bf16, 64 MiB)
//   [145MiB..)       hbuf  (T*K x H bf16, 128 MiB)
// ---------------------------------------------------------------------------
extern "C" void kernel_launch(void* const* d_in, const int* in_sizes, int n_in,
                              void* d_out, int out_size, void* d_ws, size_t ws_size,
                              hipStream_t stream) {
    (void)in_sizes; (void)n_in; (void)out_size; (void)ws_size;
    const float* x  = (const float*)d_in[0];
    const float* rw = (const float*)d_in[1];
    const float* w1 = (const float*)d_in[2];
    const float* b1 = (const float*)d_in[3];
    const float* w2 = (const float*)d_in[4];
    const float* b2 = (const float*)d_in[5];
    float* out = (float*)d_out;

    const size_t MiB = 1024ull * 1024ull;
    char* ws = (char*)d_ws;
    int*    cnt   = (int*)ws;
    int*    offs  = cnt + 8;
    float*  psum  = (float*)(offs + 8);
    int*    list  = (int*)(ws + 256);
    float*  glist = (float*)(ws + 256 + (size_t)NEXP * T_TOK * 4);
    __bf16* xb    = (__bf16*)(ws + 1 * MiB);
    __bf16* w1b   = (__bf16*)(ws + 17 * MiB);
    __bf16* w2b   = (__bf16*)(ws + 81 * MiB);
    __bf16* hbuf  = (__bf16*)(ws + 145 * MiB);

    long outN = (long)T_TOK * DDIM;
    moe_zero<<<1024, 256, 0, stream>>>(out, outN, cnt, psum);
    moe_cvt<<<2048, 256, 0, stream>>>(x,  xb,  (long)T_TOK * DDIM / 4);
    moe_cvt<<<4096, 256, 0, stream>>>(w1, w1b, (long)NEXP * DDIM * HDIM / 4);
    moe_cvt<<<4096, 256, 0, stream>>>(w2, w2b, (long)NEXP * HDIM * DDIM / 4);
    moe_router<<<T_TOK / 8, 256, 0, stream>>>(x, rw, cnt, psum, list, glist);
    moe_aux<<<1, 32, 0, stream>>>(cnt, psum, offs, out + outN);
    moe_ffn1<<<dim3(HDIM / BN, T_TOK / BM, NEXP), 256, 0, stream>>>(
        xb, w1b, b1, cnt, offs, list, hbuf);
    moe_ffn2<<<dim3(DDIM / BN, T_TOK / BM, NEXP), 256, 0, stream>>>(
        hbuf, w2b, b2, cnt, offs, list, glist, out);
}
